// MultiheadBlockAttention_4715874091643
// MI455X (gfx1250) — matrix operand order repaired and checked by reading, not compiled, not run on hardware
//
#include <hip/hip_runtime.h>

// Problem constants (reference: B,S,NB,D,H = 4,512,32,512,8)
#define BB   4
#define SSQ  512
#define NBLK 32
#define DD   512
#define HH   8
#define DHD  64

typedef __attribute__((ext_vector_type(16))) __bf16          v16bf;
typedef __attribute__((ext_vector_type(8)))  float           v8f;
typedef __attribute__((ext_vector_type(16))) unsigned short  ushort16;
typedef __attribute__((ext_vector_type(8)))  unsigned short  ushort8;

// LDS layout (bytes), dynamic shared:
//   [0,      99840)  staged inputs xq/xk/xv bf16, [32][520] each (33280 B each)
//                    -> reused after 2nd barrier as per-head P scratch [32][40] bf16
//   [99840, 133120)  Q bf16 [32][520]
//   [133120,166400)  K bf16 [32][520]
//   [166400,207360)  V^T bf16 [512][40]
//   [207360,207488)  padding mask, 32 floats
#define XP  520
#define VP  40
#define SMEM_BYTES 207488

__device__ __forceinline__ unsigned short f2bf(float f) {
  unsigned int u = __float_as_uint(f);
  u += 0x7fffu + ((u >> 16) & 1u);          // round-to-nearest-even
  return (unsigned short)(u >> 16);
}

__device__ __forceinline__ v8f zero8() {
  v8f z;
#pragma unroll
  for (int i = 0; i < 8; ++i) z[i] = 0.0f;
  return z;
}

__device__ __forceinline__ v16bf make_frag(ushort8 lo, ushort8 hi) {
  ushort16 w = __builtin_shufflevector(lo, hi,
      0,1,2,3,4,5,6,7,8,9,10,11,12,13,14,15);
  return __builtin_bit_cast(v16bf, w);
}

// A-fragment 16x32 bf16: lane (0..15) holds row M=lane, lane-half lh selects
// K = 16*lh + 0..15 contiguous (matches frag_B).
__device__ __forceinline__ v16bf frag_A(const unsigned short* rowbase, int lh) {
  ushort8 lo = *(const ushort8*)(rowbase + 16 * lh);
  ushort8 hi = *(const ushort8*)(rowbase + 16 * lh + 8);
  return make_frag(lo, hi);
}

// B-fragment 32x16 bf16 given an N-major operand (row n = column N of B):
// lane holds N = lane&15, K = 16*lh + 0..15 contiguous.
__device__ __forceinline__ v16bf frag_B(const unsigned short* rowk) {
  ushort8 lo = *(const ushort8*)(rowk);
  ushort8 hi = *(const ushort8*)(rowk + 8);
  return make_frag(lo, hi);
}

#define WMMA_BF16(A, Bf, C) \
  __builtin_amdgcn_wmma_f32_16x16x32_bf16(false, (A), false, (Bf), (short)0, (C), false, false)

// ---------------------------------------------------------------------------
// Kernel 1: convert wq|wk|wv fp32 -> bf16 into workspace (keeps [N][K] layout)
// ---------------------------------------------------------------------------
__global__ __launch_bounds__(256) void wcvt_kernel(const float* __restrict__ wq,
                                                   const float* __restrict__ wk,
                                                   const float* __restrict__ wv,
                                                   unsigned short* __restrict__ dst) {
  int i = blockIdx.x * 256 + threadIdx.x;          // 0 .. 3*512*512-1
  const float* src = (i < DD * DD) ? wq : ((i < 2 * DD * DD) ? wk : wv);
  int r = i & (DD * DD - 1);
  dst[i] = f2bf(src[r]);
}

// ---------------------------------------------------------------------------
// Kernel 2: fused QKV projection + per-(b,s) block attention.
// One workgroup per (b,s); wave w owns output dims [64w,64w+64) == head w.
// ---------------------------------------------------------------------------
__global__ __launch_bounds__(256, 1) void mba_kernel(
    const float* __restrict__ xq, const float* __restrict__ xk,
    const float* __restrict__ xv, const float* __restrict__ pmask,
    const unsigned short* __restrict__ wbf,
    const float* __restrict__ bq, const float* __restrict__ bk,
    const float* __restrict__ bv, float* __restrict__ out) {
  extern __shared__ char smem[];
  unsigned short* ldsX0 = (unsigned short*)(smem);                 // 3 tensors
  unsigned short* ldsQ  = (unsigned short*)(smem + 99840);
  unsigned short* ldsK  = (unsigned short*)(smem + 133120);
  unsigned short* ldsVT = (unsigned short*)(smem + 166400);
  float*          ldsM  = (float*)(smem + 207360);

  const int tid  = threadIdx.x;
  const int wid  = tid >> 5;          // wave id == head id
  const int lane = tid & 31;
  const int lh   = lane >> 4;         // lane half
  const int l15  = lane & 15;
  const int bs   = blockIdx.x;        // b*S + s
  const int b    = bs / SSQ;
  const int s    = bs % SSQ;

  // ---- stage inputs (fp32 -> bf16) into LDS --------------------------------
  const float* xs[3] = {xq, xk, xv};
  const size_t xoff  = (size_t)bs * (NBLK * DD);
#pragma unroll
  for (int t = 0; t < 3; ++t) {
    const float* src = xs[t] + xoff;
    unsigned short* dst = ldsX0 + t * (NBLK * XP);
    for (int i = tid; i < NBLK * DD; i += 256) {
      int n = i >> 9, kk = i & (DD - 1);
      dst[n * XP + kk] = f2bf(src[i]);
    }
  }
  if (tid < NBLK) ldsM[tid] = pmask[(size_t)bs * NBLK + tid];
  __syncthreads();

  // ---- projections: Y[32,64w..] = X[32,512] @ W^T, bf16 WMMA, f32 accum ----
  const float* biases[3] = {bq, bk, bv};
#pragma unroll
  for (int t = 0; t < 3; ++t) {
    v8f acc[2][4];
#pragma unroll
    for (int mt = 0; mt < 2; ++mt)
#pragma unroll
      for (int nt = 0; nt < 4; ++nt) acc[mt][nt] = zero8();

    const unsigned short* W    = wbf + (size_t)t * (DD * DD);
    const unsigned short* xlds = ldsX0 + t * (NBLK * XP);
    const unsigned short* arow0 = xlds + l15 * XP;
    const unsigned short* arow1 = xlds + (16 + l15) * XP;
    const unsigned short* brow[4];
#pragma unroll
    for (int nt = 0; nt < 4; ++nt)
      brow[nt] = W + (size_t)(wid * 64 + nt * 16 + l15) * DD;

    for (int ks = 0; ks < 16; ++ks) {
      const int k0 = ks * 32;
      v16bf a0 = frag_A(arow0 + k0, lh);
      v16bf a1 = frag_A(arow1 + k0, lh);
#pragma unroll
      for (int nt = 0; nt < 4; ++nt) {
        v16bf bf = frag_B(brow[nt] + k0 + 16 * lh);
        acc[0][nt] = WMMA_BF16(a0, bf, acc[0][nt]);
        acc[1][nt] = WMMA_BF16(a1, bf, acc[1][nt]);
        if (ks < 15) __builtin_prefetch(brow[nt] + k0 + 32 + 16 * lh, 0, 0);
      }
    }

    // bias add + bf16 store (Q,K row-major; V transposed)
    const float* bias = biases[t];
#pragma unroll
    for (int mt = 0; mt < 2; ++mt)
#pragma unroll
      for (int nt = 0; nt < 4; ++nt) {
        const int ncol = wid * 64 + nt * 16 + l15;
        const float bvv = bias[ncol];
#pragma unroll
        for (int i = 0; i < 8; ++i) {
          const int m = mt * 16 + i + 8 * lh;       // C layout: M = vgpr + 8*lh
          const unsigned short hv = f2bf(acc[mt][nt][i] + bvv);
          if (t == 0)      ldsQ[m * XP + ncol] = hv;
          else if (t == 1) ldsK[m * XP + ncol] = hv;
          else             ldsVT[ncol * VP + m] = hv;   // V^T for P@V B-frags
        }
      }
  }
  __syncthreads();

  // ---- attention for head h = wid -----------------------------------------
  const int h = wid;
  v8f sacc[2][2];
#pragma unroll
  for (int mt = 0; mt < 2; ++mt)
#pragma unroll
    for (int nt = 0; nt < 2; ++nt) sacc[mt][nt] = zero8();

  const unsigned short* qrow0 = ldsQ + l15 * XP;
  const unsigned short* qrow1 = ldsQ + (16 + l15) * XP;
#pragma unroll
  for (int ks = 0; ks < 2; ++ks) {
    const int d0 = h * 64 + ks * 32;
    v16bf a0 = frag_A(qrow0 + d0, lh);
    v16bf a1 = frag_A(qrow1 + d0, lh);
#pragma unroll
    for (int nt = 0; nt < 2; ++nt) {
      v16bf bf = frag_B(ldsK + (size_t)(nt * 16 + l15) * XP + d0 + 16 * lh);
      sacc[0][nt] = WMMA_BF16(a0, bf, sacc[0][nt]);
      sacc[1][nt] = WMMA_BF16(a1, bf, sacc[1][nt]);
    }
  }

  const float m0 = ldsM[l15];
  const float m1 = ldsM[16 + l15];
  float* wout = out + (size_t)BB * SSQ * NBLK * DD +
                ((((size_t)b * HH + h) * SSQ + s) * NBLK * NBLK);
  unsigned short* ldsP = (unsigned short*)(smem + h * (NBLK * VP * 2));

  // row softmax over 32 columns (two N tiles, 16-lane groups per lane half)
#pragma unroll
  for (int mt = 0; mt < 2; ++mt) {
#pragma unroll
    for (int i = 0; i < 8; ++i) {
      float v0 = sacc[mt][0][i] * 0.125f + m0;     // 1/sqrt(64)
      float v1 = sacc[mt][1][i] * 0.125f + m1;
      float mx = fmaxf(v0, v1);
      mx = fmaxf(mx, __shfl_xor(mx, 1, 32));
      mx = fmaxf(mx, __shfl_xor(mx, 2, 32));
      mx = fmaxf(mx, __shfl_xor(mx, 4, 32));
      mx = fmaxf(mx, __shfl_xor(mx, 8, 32));
      float e0 = __expf(v0 - mx);
      float e1 = __expf(v1 - mx);
      float sm = e0 + e1;
      sm += __shfl_xor(sm, 1, 32);
      sm += __shfl_xor(sm, 2, 32);
      sm += __shfl_xor(sm, 4, 32);
      sm += __shfl_xor(sm, 8, 32);
      const float inv = __builtin_amdgcn_rcpf(sm);
      const float p0 = e0 * inv, p1 = e1 * inv;
      const int nr = mt * 16 + i + 8 * lh;
      wout[nr * NBLK + l15]      = p0;             // attn_weight (fp32)
      wout[nr * NBLK + 16 + l15] = p1;
      ldsP[nr * VP + l15]        = f2bf(p0);       // P for next matmul
      ldsP[nr * VP + 16 + l15]   = f2bf(p1);
    }
  }

  // O = P(32x32) @ V(32x64); same-wave DS ops are in-order, no barrier needed
  v8f oacc[2][4];
#pragma unroll
  for (int mt = 0; mt < 2; ++mt)
#pragma unroll
    for (int nt = 0; nt < 4; ++nt) oacc[mt][nt] = zero8();

  v16bf pa0 = frag_A(ldsP + l15 * VP, lh);
  v16bf pa1 = frag_A(ldsP + (16 + l15) * VP, lh);
#pragma unroll
  for (int nt = 0; nt < 4; ++nt) {
    v16bf bf = frag_B(ldsVT + (size_t)(h * 64 + nt * 16 + l15) * VP + 16 * lh);
    oacc[0][nt] = WMMA_BF16(pa0, bf, oacc[0][nt]);
    oacc[1][nt] = WMMA_BF16(pa1, bf, oacc[1][nt]);
  }

#pragma unroll
  for (int mt = 0; mt < 2; ++mt)
#pragma unroll
    for (int nt = 0; nt < 4; ++nt) {
      const int dc = h * 64 + nt * 16 + l15;
#pragma unroll
      for (int i = 0; i < 8; ++i) {
        const int nr = mt * 16 + i + 8 * lh;
        out[((size_t)bs * NBLK + nr) * DD + dc] = oacc[mt][nt][i];
      }
    }
}

// ---------------------------------------------------------------------------
extern "C" void kernel_launch(void* const* d_in, const int* in_sizes, int n_in,
                              void* d_out, int out_size, void* d_ws, size_t ws_size,
                              hipStream_t stream) {
  const float* q    = (const float*)d_in[0];
  const float* k    = (const float*)d_in[1];
  const float* v    = (const float*)d_in[2];
  const float* mask = (const float*)d_in[3];
  const float* wq   = (const float*)d_in[4];
  const float* bq   = (const float*)d_in[5];
  const float* wk   = (const float*)d_in[6];
  const float* bk   = (const float*)d_in[7];
  const float* wv   = (const float*)d_in[8];
  const float* bv   = (const float*)d_in[9];
  unsigned short* wbf = (unsigned short*)d_ws;   // 3*512*512*2 = 1.5 MB scratch
  float* out = (float*)d_out;

  wcvt_kernel<<<(3 * DD * DD) / 256, 256, 0, stream>>>(wq, wk, wv, wbf);
  mba_kernel<<<BB * SSQ, 256, SMEM_BYTES, stream>>>(q, k, v, mask, wbf,
                                                    bq, bk, bv, out);
}
